// GAT_4166118277716
// MI455X (gfx1250) — matrix operand order repaired and checked by reading, not compiled, not run on hardware
//
#include <hip/hip_runtime.h>
#include <hip/hip_bf16.h>

// ---------------------------------------------------------------------------
// Two-layer GAT for MI455X (gfx1250, wave32).
//  - Dense linear layers: bf16 WMMA (v_wmma_f32_16x16x32_bf16), f32 accum,
//    explicit double-buffered K-loop so loads of iteration k+1 overlap the
//    4 back-to-back WMMAs of iteration k (avoids s_wait_loadcnt 0 per wmma).
//  - Edge softmax: 3-pass (atomicMax on ordered-uint, exp+atomicAdd,
//    normalize + scatter atomicAdd). h matrices fit in 192MB L2, so the
//    per-edge gathers/atomics are L2-resident.
// ---------------------------------------------------------------------------

typedef __attribute__((ext_vector_type(16))) __bf16 bf16x16;
typedef __attribute__((ext_vector_type(8)))  float  f32x8;

#define NN   20000
#define NE   320000
#define INC  128
#define HIDC 64
#define NH   8
#define OUTC 64
#define D1   (NH * HIDC)   // 512
#define NEG_SLOPE 0.2f

union ABFrag { bf16x16 v; uint4 u[2]; };
struct Frags { ABFrag a; ABFrag b[4]; };

// order-preserving float<->uint encoding for atomicMax on floats
__device__ __forceinline__ unsigned encf(float f) {
  unsigned u = __float_as_uint(f);
  return (u & 0x80000000u) ? ~u : (u | 0x80000000u);
}
__device__ __forceinline__ float decf(unsigned u) {
  unsigned v = (u & 0x80000000u) ? (u & 0x7FFFFFFFu) : ~u;
  return __uint_as_float(v);
}

// ---------------------------------------------------------------------------
// Utility kernels
// ---------------------------------------------------------------------------
__global__ __launch_bounds__(256) void zero_k(unsigned* __restrict__ p, long n) {
  long t = (long)blockIdx.x * blockDim.x + threadIdx.x;
  if (t < n) p[t] = 0u;
}

__global__ __launch_bounds__(256) void cvt_bf16_k(const float* __restrict__ in,
                                                  __bf16* __restrict__ out, long n) {
  long t = (long)blockIdx.x * blockDim.x + threadIdx.x;
  if (t < n) out[t] = (__bf16)in[t];
}

// W[K][Nn] fp32 (row major) -> Wt[Nn][K] bf16 (row major)
__global__ __launch_bounds__(256) void trcvt_k(const float* __restrict__ W,
                                               __bf16* __restrict__ Wt, int K, int Nn) {
  int t = blockIdx.x * blockDim.x + threadIdx.x;
  if (t >= K * Nn) return;
  int k = t % K, n = t / K;
  Wt[t] = (__bf16)W[(long)k * Nn + n];
}

// ---------------------------------------------------------------------------
// WMMA GEMM: C[M][Nn] = A[M][K] (bf16) x Bt[Nn][K]^T (bf16), f32 accum.
// One wave computes a 16x64 output stripe (4 N-tiles sharing the A frag).
// Requires M%16==0, Nn%64==0, K%32==0. Tile branch is wave-uniform -> EXEC
// stays all ones around every v_wmma (ISA 7.12 requirement).
// ---------------------------------------------------------------------------
__device__ __forceinline__ void load_frags(Frags& f, const __bf16* Arow,
                                           const __bf16* Bbase, int k,
                                           int aoff, int boff, int K) {
  // A layout: lane (m, half) holds 16 contiguous K values (matches B)
  f.a.u[0] = *(const uint4*)(Arow + k + aoff);
  f.a.u[1] = *(const uint4*)(Arow + k + aoff + 8);
#pragma unroll
  for (int j = 0; j < 4; ++j) {
    // B layout: lane (n, half) holds 16 contiguous K values (same order as A)
    const __bf16* bp = Bbase + (long)j * 16 * K + k + boff;
    f.b[j].u[0] = *(const uint4*)(bp);
    f.b[j].u[1] = *(const uint4*)(bp + 8);
  }
}

__global__ __launch_bounds__(256)
void wmma_gemm_k(const __bf16* __restrict__ A, const __bf16* __restrict__ Bt,
                 float* __restrict__ C, int M, int Nn, int K) {
  const int lane = threadIdx.x & 31;
  const int wid  = threadIdx.x >> 5;
  const int tilesN4 = Nn >> 6;                 // groups of 4 16-wide N tiles
  const int wt  = blockIdx.x * 8 + wid;
  const int tm  = wt / tilesN4;
  const int tn4 = wt - tm * tilesN4;
  if (tm >= (M >> 4)) return;                  // wave-uniform

  const int mlo  = lane & 15;
  const int aoff = (lane >> 4) << 4;           // 0 or 16 (A K-offset)
  const int boff = (lane >> 4) << 4;           // 0 or 16 (B K-offset)
  const __bf16* Arow  = A  + (long)(tm * 16 + mlo) * K;
  const __bf16* Bbase = Bt + (long)(tn4 * 64 + mlo) * K;

  f32x8 acc[4] = {};
  Frags cur, nxt;
  load_frags(cur, Arow, Bbase, 0, aoff, boff, K);

  for (int k = 32; k < K; k += 32) {
    // issue next iteration's 10 b128 loads before consuming the current ones
    load_frags(nxt, Arow, Bbase, k, aoff, boff, K);
    __builtin_prefetch(Arow + k + 32, 0, 3);   // global_prefetch_b8 (near)
#pragma unroll
    for (int j = 0; j < 4; ++j)
      acc[j] = __builtin_amdgcn_wmma_f32_16x16x32_bf16(
          false, cur.a.v, false, cur.b[j].v, (short)0, acc[j], false, false);
    cur = nxt;
  }
#pragma unroll
  for (int j = 0; j < 4; ++j)
    acc[j] = __builtin_amdgcn_wmma_f32_16x16x32_bf16(
        false, cur.a.v, false, cur.b[j].v, (short)0, acc[j], false, false);

  // C/D layout: VGPR r, lanes<16 -> M=r, lanes>=16 -> M=r+8; N = lane&15
  const long rbase = (long)(tm * 16 + ((lane >> 4) << 3));
#pragma unroll
  for (int j = 0; j < 4; ++j) {
    float* Cp = C + rbase * Nn + tn4 * 64 + j * 16 + mlo;
#pragma unroll
    for (int r = 0; r < 8; ++r) Cp[(long)r * Nn] = acc[j][r];
  }
}

// ---------------------------------------------------------------------------
// Attention dot products: a_s[n,h] = sum_c h[n,h,c]*att_src[h,c] (same a_d)
// ---------------------------------------------------------------------------
template <int HEADS_, int CH>
__global__ __launch_bounds__(256)
void att_dots_k(const float* __restrict__ h, const float* __restrict__ atts,
                const float* __restrict__ attd, float* __restrict__ as_,
                float* __restrict__ ad_, int n) {
  int t = blockIdx.x * blockDim.x + threadIdx.x;
  if (t >= n * HEADS_) return;
  int hd = t % HEADS_;
  const float* row = h + (long)t * CH;      // t = node*HEADS_+hd, stride HEADS_*CH
  const float* s = atts + hd * CH;
  const float* d = attd + hd * CH;
  float accs = 0.f, accd = 0.f;
#pragma unroll 4
  for (int c = 0; c < CH; c += 4) {
    float4 hv = *(const float4*)(row + c);
    float4 sv = *(const float4*)(s + c);
    float4 dv = *(const float4*)(d + c);
    accs += hv.x * sv.x + hv.y * sv.y + hv.z * sv.z + hv.w * sv.w;
    accd += hv.x * dv.x + hv.y * dv.y + hv.z * dv.z + hv.w * dv.w;
  }
  as_[t] = accs;
  ad_[t] = accd;
}

// edge i < NE comes from edge_index; i >= NE is the self loop (i-NE, i-NE)
__device__ __forceinline__ void edge_sd(const int* __restrict__ ei, int e, int& s, int& d) {
  if (e < NE) { s = ei[e]; d = ei[NE + e]; } else { s = d = e - NE; }
}

template <int HEADS_>
__global__ __launch_bounds__(256)
void edge_max_k(const int* __restrict__ ei, const float* __restrict__ as_,
                const float* __restrict__ ad_, unsigned* __restrict__ m) {
  int t = blockIdx.x * blockDim.x + threadIdx.x;
  if (t >= (NE + NN) * HEADS_) return;
  int e = t / HEADS_, hd = t % HEADS_;
  int s, d; edge_sd(ei, e, s, d);
  float v = as_[s * HEADS_ + hd] + ad_[d * HEADS_ + hd];
  v = v > 0.f ? v : NEG_SLOPE * v;                 // leaky_relu
  atomicMax(&m[d * HEADS_ + hd], encf(v));         // segment_max
}

template <int HEADS_>
__global__ __launch_bounds__(256)
void edge_expsum_k(const int* __restrict__ ei, const float* __restrict__ as_,
                   const float* __restrict__ ad_, const unsigned* __restrict__ m,
                   float* __restrict__ p, float* __restrict__ den) {
  int t = blockIdx.x * blockDim.x + threadIdx.x;
  if (t >= (NE + NN) * HEADS_) return;
  int e = t / HEADS_, hd = t % HEADS_;
  int s, d; edge_sd(ei, e, s, d);
  float v = as_[s * HEADS_ + hd] + ad_[d * HEADS_ + hd];
  v = v > 0.f ? v : NEG_SLOPE * v;
  float pe = __expf(v - decf(m[d * HEADS_ + hd]));
  p[t] = pe;
  atomicAdd(&den[d * HEADS_ + hd], pe);            // segment_sum
}

template <int HEADS_, int CH>
__global__ __launch_bounds__(256)
void edge_agg_k(const int* __restrict__ ei, const float* __restrict__ p,
                const float* __restrict__ den, const float* __restrict__ h,
                float* __restrict__ agg) {
  int t = blockIdx.x * blockDim.x + threadIdx.x;
  if (t >= (NE + NN) * HEADS_) return;
  int e = t / HEADS_, hd = t % HEADS_;
  int s, d; edge_sd(ei, e, s, d);
  float alpha = p[t] / den[d * HEADS_ + hd];
  const float* hrow = h + ((long)s * HEADS_ + hd) * CH;   // L2-resident gather
  float* arow = agg + ((long)d * HEADS_ + hd) * CH;
#pragma unroll 4
  for (int c = 0; c < CH; c += 4) {
    float4 hv = *(const float4*)(hrow + c);
    atomicAdd(arow + c + 0, hv.x * alpha);
    atomicAdd(arow + c + 1, hv.y * alpha);
    atomicAdd(arow + c + 2, hv.z * alpha);
    atomicAdd(arow + c + 3, hv.w * alpha);
  }
}

// h_bf16 = bf16(relu(agg + b[col]))  -- layer-1 epilogue + GEMM2 input prep
__global__ __launch_bounds__(256)
void relu_bias_cvt_k(const float* __restrict__ agg, const float* __restrict__ b,
                     __bf16* __restrict__ out, long n, int cols) {
  long t = (long)blockIdx.x * blockDim.x + threadIdx.x;
  if (t >= n) return;
  float v = agg[t] + b[t % cols];
  out[t] = (__bf16)(v > 0.f ? v : 0.f);
}

__global__ __launch_bounds__(256)
void add_bias_k(float* __restrict__ out, const float* __restrict__ b, long n, int cols) {
  long t = (long)blockIdx.x * blockDim.x + threadIdx.x;
  if (t >= n) return;
  out[t] += b[t % cols];
}

// ---------------------------------------------------------------------------
// Launch
// ---------------------------------------------------------------------------
extern "C" void kernel_launch(void* const* d_in, const int* in_sizes, int n_in,
                              void* d_out, int out_size, void* d_ws, size_t ws_size,
                              hipStream_t stream) {
  (void)in_sizes; (void)n_in; (void)out_size; (void)ws_size;

  const float* x    = (const float*)d_in[0];
  const int*   ei   = (const int*)d_in[1];
  const float* W1   = (const float*)d_in[2];
  const float* atS1 = (const float*)d_in[3];
  const float* atD1 = (const float*)d_in[4];
  const float* b1   = (const float*)d_in[5];
  const float* W2   = (const float*)d_in[6];
  const float* atS2 = (const float*)d_in[7];
  const float* atD2 = (const float*)d_in[8];
  const float* b2   = (const float*)d_in[9];
  float* out = (float*)d_out;

  // linear workspace layout (256B aligned), with deliberate buffer reuse
  char* w = (char*)d_ws;
  auto take = [&](size_t bytes) { char* p = w; w += (bytes + 255) & ~(size_t)255; return p; };
  __bf16*   xb   = (__bf16*)take((size_t)NN * INC * 2);        // later: h2 (fp32, same size)
  __bf16*   W1t  = (__bf16*)take((size_t)D1 * INC * 2);
  __bf16*   W2t  = (__bf16*)take((size_t)OUTC * D1 * 2);
  float*    h1   = (float*)take((size_t)NN * D1 * 4);          // later: hb (bf16, half size)
  float*    aS1  = (float*)take((size_t)NN * NH * 4);
  float*    aD1  = (float*)take((size_t)NN * NH * 4);
  unsigned* m1   = (unsigned*)take((size_t)NN * NH * 4);
  float*    den1 = (float*)take((size_t)NN * NH * 4);
  float*    p1   = (float*)take((size_t)(NE + NN) * NH * 4);   // later: p2
  float*    agg1 = (float*)take((size_t)NN * D1 * 4);
  float*    aS2  = (float*)take((size_t)NN * 4);
  float*    aD2  = (float*)take((size_t)NN * 4);
  unsigned* m2   = (unsigned*)take((size_t)NN * 4);
  float*    den2 = (float*)take((size_t)NN * 4);

  __bf16* hb = (__bf16*)h1;   // h1 dead after layer-1 aggregation
  float*  h2 = (float*)xb;    // xb dead after GEMM1
  float*  p2 = p1;            // p1 dead after layer-1 aggregation

  const int B = 256;
  auto blocks = [](long n) { return (int)((n + 255) / 256); };

  // ---------------- Layer 1 ----------------
  cvt_bf16_k<<<blocks((long)NN * INC), B, 0, stream>>>(x, xb, (long)NN * INC);
  trcvt_k<<<blocks(INC * D1), B, 0, stream>>>(W1, W1t, INC, D1);
  trcvt_k<<<blocks(D1 * OUTC), B, 0, stream>>>(W2, W2t, D1, OUTC);

  { // h1 = x @ W1  (20000x512, K=128)
    int waves = (NN / 16) * (D1 / 64);      // 10000 waves, 16x64 stripe each
    wmma_gemm_k<<<(waves + 7) / 8, B, 0, stream>>>(xb, W1t, h1, NN, D1, INC);
  }
  att_dots_k<NH, HIDC><<<blocks((long)NN * NH), B, 0, stream>>>(h1, atS1, atD1, aS1, aD1, NN);

  zero_k<<<blocks((long)NN * NH), B, 0, stream>>>(m1, (long)NN * NH);
  zero_k<<<blocks((long)NN * NH), B, 0, stream>>>((unsigned*)den1, (long)NN * NH);
  zero_k<<<blocks((long)NN * D1), B, 0, stream>>>((unsigned*)agg1, (long)NN * D1);

  edge_max_k<NH><<<blocks((long)(NE + NN) * NH), B, 0, stream>>>(ei, aS1, aD1, m1);
  edge_expsum_k<NH><<<blocks((long)(NE + NN) * NH), B, 0, stream>>>(ei, aS1, aD1, m1, p1, den1);
  edge_agg_k<NH, HIDC><<<blocks((long)(NE + NN) * NH), B, 0, stream>>>(ei, p1, den1, h1, agg1);

  relu_bias_cvt_k<<<blocks((long)NN * D1), B, 0, stream>>>(agg1, b1, hb, (long)NN * D1, D1);

  // ---------------- Layer 2 ----------------
  { // h2 = relu(h) @ W2  (20000x64, K=512)
    int waves = (NN / 16) * (OUTC / 64);    // 1250 waves
    wmma_gemm_k<<<(waves + 7) / 8, B, 0, stream>>>(hb, W2t, h2, NN, OUTC, D1);
  }
  att_dots_k<1, OUTC><<<blocks((long)NN), B, 0, stream>>>(h2, atS2, atD2, aS2, aD2, NN);

  zero_k<<<blocks((long)NN), B, 0, stream>>>(m2, (long)NN);
  zero_k<<<blocks((long)NN), B, 0, stream>>>((unsigned*)den2, (long)NN);
  zero_k<<<blocks((long)NN * OUTC), B, 0, stream>>>((unsigned*)out, (long)NN * OUTC);

  edge_max_k<1><<<blocks((long)(NE + NN)), B, 0, stream>>>(ei, aS2, aD2, m2);
  edge_expsum_k<1><<<blocks((long)(NE + NN)), B, 0, stream>>>(ei, aS2, aD2, m2, p2, den2);
  edge_agg_k<1, OUTC><<<blocks((long)(NE + NN)), B, 0, stream>>>(ei, p2, den2, h2, out);

  add_bias_k<<<blocks((long)NN * OUTC), B, 0, stream>>>(out, b2, (long)NN * OUTC, OUTC);
}
